// GNNActor_23192823398472
// MI455X (gfx1250) — matrix operand order repaired and checked by reading, not compiled, not run on hardware
//
#include <hip/hip_runtime.h>
#include <stdint.h>

// ---------------------------------------------------------------------------
// Types for CDNA5 WMMA (wave32): v_wmma_f32_16x16x32_bf16
// ---------------------------------------------------------------------------
typedef __attribute__((ext_vector_type(16))) __bf16 vbf16x16;
typedef __attribute__((ext_vector_type(8)))  float  v8f;

union FragBF {
    uint32_t  u[8];   // 8 dwords = 16 bf16
    vbf16x16  v;
};

#define NNODES 100000
#define NEDGES 800000
#define DF     256

// ---------------------------------------------------------------------------
// helpers
// ---------------------------------------------------------------------------
__device__ __forceinline__ uint16_t f2bf(float f) {
    uint32_t u = __float_as_uint(f);
    uint32_t r = (u >> 16) & 1u;
    u += 0x7FFFu + r;               // round-to-nearest-even
    return (uint16_t)(u >> 16);
}
__device__ __forceinline__ uint32_t pack2bf(float lo, float hi) {
    return (uint32_t)f2bf(lo) | ((uint32_t)f2bf(hi) << 16);
}
__device__ __forceinline__ v8f wmma_bf16(const FragBF& a, const FragBF& b, v8f c) {
    return __builtin_amdgcn_wmma_f32_16x16x32_bf16(
        false, a.v, false, b.v, (short)0, c, false, false);
}

// ---------------------------------------------------------------------------
// k0: zero fp32 buffer
// ---------------------------------------------------------------------------
__global__ void zero_f32(float* __restrict__ p, int n) {
    int i = blockIdx.x * blockDim.x + threadIdx.x;
    if (i < n) p[i] = 0.0f;
}

// ---------------------------------------------------------------------------
// k1: fp32 -> packed bf16 pairs
// ---------------------------------------------------------------------------
__global__ void cvt_bf16(const float* __restrict__ src, uint32_t* __restrict__ dst,
                         int n_pairs) {
    int i = blockIdx.x * blockDim.x + threadIdx.x;
    if (i < n_pairs) {
        float2 f = ((const float2*)src)[i];
        dst[i] = pack2bf(f.x, f.y);
    }
}

// ---------------------------------------------------------------------------
// k2: edge scatter -- mean-aggregation numerator + degree.
// 4 edges / block, 64 lanes / edge, float4 per lane. agg fits in 192MB L2,
// so the 205M atomic f32 adds stay on-chip.
// ---------------------------------------------------------------------------
__global__ void __launch_bounds__(256) scatter_kernel(
    const float* __restrict__ x, const long long* __restrict__ ei,
    float* __restrict__ agg, int* __restrict__ cnt, int E)
{
    int e = blockIdx.x * 4 + (threadIdx.x >> 6);
    if (e >= E) return;
    int lane = threadIdx.x & 63;           // 64 lanes * 4 floats = 256
    long long src = ei[e];
    long long dst = ei[(size_t)E + e];
    float4 v = ((const float4*)(x + (size_t)src * DF))[lane];
    float* a = agg + (size_t)dst * DF + lane * 4;
    atomicAdd(a + 0, v.x);
    atomicAdd(a + 1, v.y);
    atomicAdd(a + 2, v.z);
    atomicAdd(a + 3, v.w);
    if (lane == 0) atomicAdd(cnt + dst, 1);
}

// ---------------------------------------------------------------------------
// k3: mean = agg / max(cnt,1) -> packed bf16
// ---------------------------------------------------------------------------
__global__ void mean_kernel(const float* __restrict__ agg, const int* __restrict__ cnt,
                            uint32_t* __restrict__ meanbf, int n_pairs) {
    int i = blockIdx.x * blockDim.x + threadIdx.x;
    if (i < n_pairs) {
        int node = i >> 7;                  // 128 pairs per node
        float c = (float)cnt[node];
        c = c < 1.0f ? 1.0f : c;
        float inv = 1.0f / c;
        float2 a = ((const float2*)agg)[i];
        meanbf[i] = pack2bf(a.x * inv, a.y * inv);
    }
}

// ---------------------------------------------------------------------------
// k4: fused SAGEConv GEMM:
//   h0 = relu(mean @ Wl^T + bl + x @ Wr^T) + x         [N, 256] -> bf16
// Block: 16 rows x 256 cols. 8 waves, wave w -> cols [32w, 32w+32) (2 accs).
// K loop: 32 per step, 2 WMMAs per accumulator per step (mean & x paths).
// ---------------------------------------------------------------------------
__global__ void __launch_bounds__(256) conv_kernel(
    const uint32_t* __restrict__ xbf, const uint32_t* __restrict__ meanbf,
    const uint32_t* __restrict__ Wl, const uint32_t* __restrict__ Wr,
    const float* __restrict__ bl, const float* __restrict__ xf,
    uint32_t* __restrict__ h0bf)
{
    __shared__ __align__(16) uint32_t sMean[16 * 128];   // 16 rows x 256 bf16
    __shared__ __align__(16) uint32_t sX[16 * 128];

    const int tid  = threadIdx.x;
    const int row0 = blockIdx.x * 16;

    for (int i = tid; i < 16 * 128; i += 256) {
        int r = i >> 7, c = i & 127;
        size_t g = (size_t)(row0 + r) * 128 + c;
        sMean[i] = meanbf[g];
        sX[i]    = xbf[g];
    }
    __syncthreads();

    const int wave = tid >> 5;
    const int lane = tid & 31;
    const int lm   = lane & 15;     // A: M row / B,C: N col
    const int lh   = lane >> 4;     // half select
    const int n0   = wave * 32;

    v8f acc0 = {};
    v8f acc1 = {};

    for (int k0 = 0; k0 < DF; k0 += 32) {
        // A fragments (16-bit A layout: regs0-3 K=8h.., regs4-7 K=16+8h..)
        FragBF am, ax;
        const int ka = (k0 >> 1) + 4 * lh;       // dword idx, 16B aligned
        const int kb = ((k0 + 16) >> 1) + 4 * lh;
        const uint32_t* bm = sMean + lm * 128;
        const uint32_t* bx = sX + lm * 128;
        *(uint4*)&am.u[0] = *(const uint4*)(bm + ka);
        *(uint4*)&am.u[4] = *(const uint4*)(bm + kb);
        *(uint4*)&ax.u[0] = *(const uint4*)(bx + ka);
        *(uint4*)&ax.u[4] = *(const uint4*)(bx + kb);

        // B fragments: B[k][n] = W[n][k]; regs0-3 K=8h.., regs4-7 K=16+8h..
        FragBF bl0, bl1, br0, br1;
        const int kw = (k0 >> 1) + 4 * lh;
        const uint32_t* p;
        p = Wl + (size_t)(n0 + lm) * 128 + kw;
        *(uint4*)&bl0.u[0] = *(const uint4*)p;  *(uint4*)&bl0.u[4] = *(const uint4*)(p + 8);
        p = Wl + (size_t)(n0 + 16 + lm) * 128 + kw;
        *(uint4*)&bl1.u[0] = *(const uint4*)p;  *(uint4*)&bl1.u[4] = *(const uint4*)(p + 8);
        p = Wr + (size_t)(n0 + lm) * 128 + kw;
        *(uint4*)&br0.u[0] = *(const uint4*)p;  *(uint4*)&br0.u[4] = *(const uint4*)(p + 8);
        p = Wr + (size_t)(n0 + 16 + lm) * 128 + kw;
        *(uint4*)&br1.u[0] = *(const uint4*)p;  *(uint4*)&br1.u[4] = *(const uint4*)(p + 8);

        acc0 = wmma_bf16(am, bl0, acc0);
        acc0 = wmma_bf16(ax, br0, acc0);
        acc1 = wmma_bf16(am, bl1, acc1);
        acc1 = wmma_bf16(ax, br1, acc1);
    }

    // epilogue: + bias, relu, + residual x, store bf16
    const int   c0 = n0 + lm;
    const int   c1 = n0 + 16 + lm;
    const float bias0 = bl[c0];
    const float bias1 = bl[c1];
    uint16_t* h0 = (uint16_t*)h0bf;
    #pragma unroll
    for (int r = 0; r < 8; ++r) {
        int row = row0 + 8 * lh + r;
        float v0 = acc0[r] + bias0;  v0 = v0 > 0.0f ? v0 : 0.0f;
        float v1 = acc1[r] + bias1;  v1 = v1 > 0.0f ? v1 : 0.0f;
        v0 += xf[(size_t)row * DF + c0];
        v1 += xf[(size_t)row * DF + c1];
        h0[(size_t)row * DF + c0] = f2bf(v0);
        h0[(size_t)row * DF + c1] = f2bf(v1);
    }
}

// ---------------------------------------------------------------------------
// k5: fused MLP: relu(h0@W1^T+b1) -> relu(@W2^T+b2) -> @W3^T+b3
// One block per 16 rows; intermediates live in LDS.
// ---------------------------------------------------------------------------
__global__ void __launch_bounds__(256) mlp_kernel(
    const uint32_t* __restrict__ h0bf,
    const uint32_t* __restrict__ W1, const float* __restrict__ b1,
    const uint32_t* __restrict__ W2, const float* __restrict__ b2,
    const float* __restrict__ W3, const float* __restrict__ b3,
    float* __restrict__ out)
{
    __shared__ __align__(16) uint32_t sH0[16 * 128];  // 16 x 256 bf16
    __shared__ __align__(16) uint32_t sH1[16 * 64];   // 16 x 128 bf16
    __shared__ __align__(16) float    sH2[16 * 32];   // 16 x 32  f32

    const int tid  = threadIdx.x;
    const int row0 = blockIdx.x * 16;
    for (int i = tid; i < 16 * 128; i += 256)
        sH0[i] = h0bf[(size_t)row0 * 128 + i];
    __syncthreads();

    const int wave = tid >> 5;
    const int lane = tid & 31;
    const int lm   = lane & 15;
    const int lh   = lane >> 4;

    // ----- phase 1: h1[16x128] = relu(h0 @ W1^T + b1); wave -> 16-col tile
    {
        const int n0 = wave * 16;
        v8f acc = {};
        for (int k0 = 0; k0 < 256; k0 += 32) {
            FragBF a, b;
            const uint32_t* base = sH0 + lm * 128;
            *(uint4*)&a.u[0] = *(const uint4*)(base + (k0 >> 1) + 4 * lh);
            *(uint4*)&a.u[4] = *(const uint4*)(base + ((k0 + 16) >> 1) + 4 * lh);
            const uint32_t* p = W1 + (size_t)(n0 + lm) * 128 + (k0 >> 1) + 4 * lh;
            *(uint4*)&b.u[0] = *(const uint4*)p;
            *(uint4*)&b.u[4] = *(const uint4*)(p + 8);
            acc = wmma_bf16(a, b, acc);
        }
        const float bias = b1[n0 + lm];
        uint16_t* s1 = (uint16_t*)sH1;
        #pragma unroll
        for (int r = 0; r < 8; ++r) {
            float v = acc[r] + bias;
            v = v > 0.0f ? v : 0.0f;
            s1[(8 * lh + r) * 128 + n0 + lm] = f2bf(v);
        }
    }
    __syncthreads();

    // ----- phase 2: h2[16x32] = relu(h1 @ W2^T + b2); waves 0,1
    if (wave < 2) {
        const int n0 = wave * 16;
        v8f acc = {};
        for (int k0 = 0; k0 < 128; k0 += 32) {
            FragBF a, b;
            const uint32_t* base = sH1 + lm * 64;
            *(uint4*)&a.u[0] = *(const uint4*)(base + (k0 >> 1) + 4 * lh);
            *(uint4*)&a.u[4] = *(const uint4*)(base + ((k0 + 16) >> 1) + 4 * lh);
            const uint32_t* p = W2 + (size_t)(n0 + lm) * 64 + (k0 >> 1) + 4 * lh;
            *(uint4*)&b.u[0] = *(const uint4*)p;
            *(uint4*)&b.u[4] = *(const uint4*)(p + 8);
            acc = wmma_bf16(a, b, acc);
        }
        const float bias = b2[n0 + lm];
        #pragma unroll
        for (int r = 0; r < 8; ++r) {
            float v = acc[r] + bias;
            v = v > 0.0f ? v : 0.0f;
            sH2[(8 * lh + r) * 32 + n0 + lm] = v;
        }
    }
    __syncthreads();

    // ----- phase 3: out[row] = h2[row] . W3 + b3
    if (tid < 16) {
        float s = b3[0];
        #pragma unroll
        for (int k = 0; k < 32; ++k) s += sH2[tid * 32 + k] * W3[k];
        out[row0 + tid] = s;
    }
}

// ---------------------------------------------------------------------------
// launch
// ---------------------------------------------------------------------------
extern "C" void kernel_launch(void* const* d_in, const int* in_sizes, int n_in,
                              void* d_out, int out_size, void* d_ws, size_t ws_size,
                              hipStream_t stream) {
    const float*     x   = (const float*)d_in[0];
    const long long* ei  = (const long long*)d_in[1];
    const float*     W_l = (const float*)d_in[2];
    const float*     b_l = (const float*)d_in[3];
    const float*     W_r = (const float*)d_in[4];
    const float*     W1  = (const float*)d_in[5];
    const float*     b1  = (const float*)d_in[6];
    const float*     W2  = (const float*)d_in[7];
    const float*     b2  = (const float*)d_in[8];
    const float*     W3  = (const float*)d_in[9];
    const float*     b3  = (const float*)d_in[10];
    float* out = (float*)d_out;

    const int N = in_sizes[0] / DF;      // 100000
    const int E = in_sizes[1] / 2;       // 800000

    // workspace carve-up (256B aligned)
    uint8_t* ws = (uint8_t*)d_ws;
    size_t o = 0;
    auto carve = [&](size_t bytes) -> void* {
        void* p = ws + o;
        o = (o + bytes + 255) & ~(size_t)255;
        return p;
    };
    float*    agg    = (float*)   carve((size_t)N * DF * 4);
    int*      cnt    = (int*)     carve((size_t)N * 4);
    uint32_t* xbf    = (uint32_t*)carve((size_t)N * DF * 2);
    uint32_t* meanbf = (uint32_t*)carve((size_t)N * DF * 2);
    uint32_t* h0bf   = (uint32_t*)carve((size_t)N * DF * 2);
    uint32_t* Wl_bf  = (uint32_t*)carve((size_t)256 * DF * 2);
    uint32_t* Wr_bf  = (uint32_t*)carve((size_t)256 * DF * 2);
    uint32_t* W1_bf  = (uint32_t*)carve((size_t)128 * DF * 2);
    uint32_t* W2_bf  = (uint32_t*)carve((size_t)32 * 128 * 2);
    (void)ws_size;

    const int T = 256;
    // zero accumulators (cnt is int; all-zero bits OK)
    {
        int n = N * DF;
        zero_f32<<<(n + T - 1) / T, T, 0, stream>>>(agg, n);
        zero_f32<<<(N + T - 1) / T, T, 0, stream>>>((float*)cnt, N);
    }
    // bf16 conversions
    {
        int np;
        np = N * DF / 2;   cvt_bf16<<<(np + T - 1) / T, T, 0, stream>>>(x,   xbf,   np);
        np = 256 * DF / 2; cvt_bf16<<<(np + T - 1) / T, T, 0, stream>>>(W_l, Wl_bf, np);
        np = 256 * DF / 2; cvt_bf16<<<(np + T - 1) / T, T, 0, stream>>>(W_r, Wr_bf, np);
        np = 128 * DF / 2; cvt_bf16<<<(np + T - 1) / T, T, 0, stream>>>(W1,  W1_bf, np);
        np = 32 * 128 / 2; cvt_bf16<<<(np + T - 1) / T, T, 0, stream>>>(W2,  W2_bf, np);
    }
    // scatter (4 edges per block)
    scatter_kernel<<<(E + 3) / 4, T, 0, stream>>>(x, ei, agg, cnt, E);
    // mean
    {
        int np = N * DF / 2;
        mean_kernel<<<(np + T - 1) / T, T, 0, stream>>>(agg, cnt, meanbf, np);
    }
    // SAGEConv GEMM + relu + residual
    conv_kernel<<<N / 16, T, 0, stream>>>(xbf, meanbf, Wl_bf, Wr_bf, b_l, x, h0bf);
    // fused MLP
    mlp_kernel<<<N / 16, T, 0, stream>>>(h0bf, W1_bf, b1, W2_bf, b2, W3, b3, out);

    (void)out_size; (void)n_in;
}
